// CrossAttention_11647951307030
// MI455X (gfx1250) — matrix operand order repaired and checked by reading, not compiled, not run on hardware
//
#include <hip/hip_runtime.h>
#include <hip/hip_bf16.h>

typedef _Float16 h8  __attribute__((ext_vector_type(8)));
typedef _Float16 h16 __attribute__((ext_vector_type(16)));
typedef float    f8  __attribute__((ext_vector_type(8)));
typedef int      i4  __attribute__((ext_vector_type(4)));

#define DIM    128
#define HEADS  4
#define DH     32
#define LQ     2500
#define LQP    2512     // padded to multiple of 16
#define LK     10080    // 6*28*60, multiple of 32
#define HW_K   1680     // 28*60
#define NTOK_K 6
#define EPS    1e-5f
#define QSCALE 0.17677669529663687f  // 32^-0.5
#define NSPLIT 5
#define CPS    (LK / 32 / NSPLIT)    // 63 chunks of 32 keys per split

__device__ __forceinline__ h16 cat8(h8 lo, h8 hi) {
    return __builtin_shufflevector(lo, hi, 0,1,2,3,4,5,6,7,8,9,10,11,12,13,14,15);
}
__device__ __forceinline__ f8 wmma16(h16 a, h16 b, f8 c) {
    return __builtin_amdgcn_wmma_f32_16x16x32_f16(false, a, false, b, (short)0, c, false, false);
}
// A operand: lane m=l%16 is the row; 16 contiguous halves at K = 16g..
__device__ __forceinline__ h16 load_a16(const _Float16* row_kbase, int g) {
    h8 lo = *(const h8*)(row_kbase + g*16);
    h8 hi = *(const h8*)(row_kbase + g*16 + 8);
    return cat8(lo, hi);
}
// B operand: lane n=l%16 is the column; 16 contiguous halves at K = 16g..
__device__ __forceinline__ h16 load_b16(const _Float16* col_kbase) {
    h8 lo = *(const h8*)(col_kbase);
    h8 hi = *(const h8*)(col_kbase + 8);
    return cat8(lo, hi);
}
__device__ __forceinline__ void lds_fence() {
    __builtin_amdgcn_wave_barrier();
    asm volatile("s_wait_dscnt 0x0" ::: "memory");
}

// ---------------- Kernel 0: transpose weights (in,out)->f16 [out][in] ----------------
__global__ void prep_weights(const float* __restrict__ W, _Float16* __restrict__ WT,
                             int K, int N) {
    int i = blockIdx.x * blockDim.x + threadIdx.x;
    if (i >= N * K) return;
    int n = i / K, k = i % K;
    WT[(long)n * K + k] = (_Float16)W[(long)k * N + n];
}

// ---------------- Kernel 1: LayerNorm + 128x128 projection via WMMA ----------------
// mode 0: Q -> qh [b][head][LQP][32] (scaled by QSCALE)
// mode 1: K -> kh [b][head][LK][32]
// mode 2: V -> vT [b][head][32][LK]
__global__ void __launch_bounds__(32)
ln_proj(const float* __restrict__ x, const float* __restrict__ lnw,
        const float* __restrict__ lnb, const _Float16* __restrict__ WT,
        const float* __restrict__ bias, _Float16* __restrict__ out,
        int Lsrc, int Ldst, int mode)
{
    const int tile = blockIdx.x, b = blockIdx.y;
    const int lane = threadIdx.x, m = lane & 15, g = lane >> 4;
    const int row = tile * 16 + m;
    __shared__ _Float16 xt[16][128];

    float vals[64];
    float s = 0.f, sq = 0.f;
    if (mode == 0) {
        const bool ok = row < Lsrc;
        #pragma unroll
        for (int j = 0; j < 64; ++j) {
            int c = g * 64 + j;
            float vv = ok ? x[((long)b * DIM + c) * Lsrc + row] : 0.f;
            vals[j] = vv; s += vv; sq += vv * vv;
        }
    } else {
        const int n = row / HW_K, r = row % HW_K;
        #pragma unroll
        for (int j = 0; j < 64; ++j) {
            int c = g * 64 + j;
            float vv = x[(((long)b * NTOK_K + n) * DIM + c) * HW_K + r];
            vals[j] = vv; s += vv; sq += vv * vv;
        }
    }
    s  += __shfl_xor(s, 16, 32);
    sq += __shfl_xor(sq, 16, 32);
    const float mean = s * (1.f / 128.f);
    const float var  = sq * (1.f / 128.f) - mean * mean;
    const float rstd = rsqrtf(var + EPS);
    #pragma unroll
    for (int j = 0; j < 64; ++j) {
        int c = g * 64 + j;
        xt[m][c] = (_Float16)((vals[j] - mean) * rstd * lnw[c] + lnb[c]);
    }
    lds_fence();

    const float scale = (mode == 0) ? QSCALE : 1.0f;
    #pragma unroll
    for (int t = 0; t < 8; ++t) {           // output channel tile, n = t*16 + m
        const int n = t * 16 + m;
        f8 acc = {};
        const _Float16* wrow = WT + (long)n * DIM;
        #pragma unroll
        for (int kc = 0; kc < 4; ++kc) {
            h16 a = load_a16(&xt[m][kc * 32], g);
            h16 bb = load_b16(wrow + kc * 32 + g * 16);
            acc = wmma16(a, bb, acc);
        }
        const float bn = bias[n];
        if (mode == 2) {                     // vT: [b][head][dh][LK], 8 consecutive tokens
            h8 pack;
            #pragma unroll
            for (int v = 0; v < 8; ++v) pack[v] = (_Float16)(acc[v] + bn);
            long o = (((long)b * HEADS + (n >> 5)) * DH + (n & 31)) * (long)Ldst
                     + tile * 16 + g * 8;
            *(h8*)(out + o) = pack;
        } else {                             // qh/kh: [b][head][L][32]
            #pragma unroll
            for (int v = 0; v < 8; ++v) {
                int rr = tile * 16 + v + g * 8;
                long o = (((long)b * HEADS + (n >> 5)) * (long)Ldst + rr) * DH + (n & 31);
                out[o] = (_Float16)((acc[v] + bn) * scale);
            }
        }
    }
}

// ---------------- Kernel 2: split-K flash attention, transposed score form -----------
// S^T = K @ Q^T  (A = K rows, B = Q cols)  -> per-lane scalar softmax stats
// O^T = V^T @ P  (A = V^T rows, B = P from C/D via one xor-16 exchange)  -> no LDS
__global__ void __launch_bounds__(32)
attn(const _Float16* __restrict__ qh, const _Float16* __restrict__ kh,
     const _Float16* __restrict__ vT,
     float* __restrict__ po, float* __restrict__ pm, float* __restrict__ pl)
{
    const int qt = blockIdx.x, head = blockIdx.y;
    const int b = blockIdx.z & 1, sp = blockIdx.z >> 1;
    const int lane = threadIdx.x, m = lane & 15, g = lane >> 4;

    const long bh = (long)b * HEADS + head;
    // Q as B operand: N = query = m, K(dh) = h + 16g  (32B contiguous per lane)
    const h16 bq = load_b16(qh + (bh * LQP + qt * 16 + m) * DH + g * 16);
    const _Float16* kbase = kh + bh * (long)LK * DH;
    const _Float16* vbase = vT + bh * (long)DH * LK;

    f8 o0 = {}, o1 = {};                 // O^T: col = query m, rows = dh v+8g (+16)
    float mrun = -1e30f, lrun = 0.f;

    for (int kc = sp * CPS; kc < (sp + 1) * CPS; ++kc) {
        const int k0 = kc * 32;
        const f8 zc = {};
        // S^T tiles: lane holds one query column; keys spread over v and xor-16 partner
        f8 st0 = wmma16(load_a16(kbase + (long)(k0 + m) * DH, g), bq, zc);
        f8 st1 = wmma16(load_a16(kbase + (long)(k0 + 16 + m) * DH, g), bq, zc);

        float mx = st0[0];
        #pragma unroll
        for (int v = 1; v < 8; ++v) mx = fmaxf(mx, st0[v]);
        #pragma unroll
        for (int v = 0; v < 8; ++v) mx = fmaxf(mx, st1[v]);
        mx = fmaxf(mx, __shfl_xor(mx, 16, 32));
        const float mn = fmaxf(mrun, mx);
        const float corr = __expf(mrun - mn);
        mrun = mn;

        float rs = 0.f;
        h8 p0h, p1h;
        #pragma unroll
        for (int v = 0; v < 8; ++v) {
            const float p0 = __expf(st0[v] - mn);
            const float p1 = __expf(st1[v] - mn);
            rs += p0 + p1;
            p0h[v] = (_Float16)p0;
            p1h[v] = (_Float16)p1;
        }
        rs += __shfl_xor(rs, 16, 32);
        lrun = lrun * corr + rs;
        #pragma unroll
        for (int v = 0; v < 8; ++v) { o0[v] *= corr; o1[v] *= corr; }

        // Build P as B operand: g=0 lanes need keys 0..15 (tile0), g=1 keys 16..31 (tile1);
        // missing 8 keys live in the xor-16 partner lane.
        i4 x0 = __builtin_bit_cast(i4, p0h);
        i4 x1 = __builtin_bit_cast(i4, p1h);
        #pragma unroll
        for (int j = 0; j < 4; ++j) {
            x0[j] = __shfl_xor(x0[j], 16, 32);
            x1[j] = __shfl_xor(x1[j], 16, 32);
        }
        const h8 p0o = __builtin_bit_cast(h8, x0);
        const h8 p1o = __builtin_bit_cast(h8, x1);
        const h16 bp = g ? cat8(p1o, p1h) : cat8(p0h, p0o);

        // O^T += V^T @ P   (A rows = dh, K = keys)
        o0 = wmma16(load_a16(vbase + (long)m * LK + k0, g), bp, o0);
        o1 = wmma16(load_a16(vbase + (long)(16 + m) * LK + k0, g), bp, o1);
    }

    // partial store: lane has fixed token, dh contiguous over v -> 32B vector stores
    const int tok = qt * 16 + m;
    const long base = (((long)sp * 2 + b) * HEADS + head) * LQP + tok;
    *(f8*)(po + base * DH + g * 8)      = o0;   // dh = g*8 + v
    *(f8*)(po + base * DH + 16 + g * 8) = o1;   // dh = 16 + g*8 + v
    if (g == 0) { pm[base] = mrun; pl[base] = lrun; }
}

// ---------------- Kernel 2b: log-sum-exp combine of split-K partials -----------------
__global__ void combine(const float* __restrict__ po, const float* __restrict__ pm,
                        const float* __restrict__ pl, float* __restrict__ aout)
{
    const int id = blockIdx.x * blockDim.x + threadIdx.x;
    if (id >= 2 * LQP * DIM) return;
    const int c = id & (DIM - 1);
    const int tok = (id >> 7) % LQP;
    const int b = id / (LQP * DIM);
    const int head = c >> 5, dh = c & 31;
    float M = -1e30f;
    #pragma unroll
    for (int s = 0; s < NSPLIT; ++s)
        M = fmaxf(M, pm[(((long)s * 2 + b) * HEADS + head) * LQP + tok]);
    float L = 0.f, O = 0.f;
    #pragma unroll
    for (int s = 0; s < NSPLIT; ++s) {
        const long r = (((long)s * 2 + b) * HEADS + head) * LQP + tok;
        const float w = __expf(pm[r] - M);
        L += pl[r] * w;
        O += po[r * DH + dh] * w;
    }
    aout[((long)b * LQP + tok) * DIM + c] = O / L;
}

// ---------------- Kernel 3: Wp proj + prenorm + GELU MLP + residual + postnorm --------
__global__ void __launch_bounds__(32)
ffn(const float* __restrict__ a,
    const _Float16* __restrict__ WpT, const float* __restrict__ bp,
    const _Float16* __restrict__ W1T, const float* __restrict__ b1,
    const _Float16* __restrict__ W2T, const float* __restrict__ b2,
    const float* __restrict__ prew, const float* __restrict__ preb,
    const float* __restrict__ postw, const float* __restrict__ postb,
    float* __restrict__ outp)
{
    const int tile = blockIdx.x, b = blockIdx.y;
    const int lane = threadIdx.x, m = lane & 15, g = lane >> 4;
    __shared__ _Float16 xt[16][128];
    __shared__ _Float16 ht[16][256];

    const float* arow = a + ((long)b * LQP + tile * 16 + m) * DIM;
    #pragma unroll
    for (int j = 0; j < 64; ++j) {
        int c = g * 64 + j;
        xt[m][c] = (_Float16)arow[c];
    }
    lds_fence();

    // z = a @ Wp + bp
    f8 z[8];
    #pragma unroll
    for (int t = 0; t < 8; ++t) {
        const int n = t * 16 + m;
        f8 acc = {};
        const _Float16* wr = WpT + (long)n * DIM;
        #pragma unroll
        for (int kc = 0; kc < 4; ++kc)
            acc = wmma16(load_a16(&xt[m][kc * 32], g), load_b16(wr + kc * 32 + g * 16), acc);
        const float bn = bp[n];
        #pragma unroll
        for (int v = 0; v < 8; ++v) acc[v] += bn;
        z[t] = acc;
    }
    // prenorm (rows = tokens v+8g, cols spread over tiles t and lanes)
    float mean[8], rstd[8];
    #pragma unroll
    for (int v = 0; v < 8; ++v) {
        float s = 0.f, sq = 0.f;
        #pragma unroll
        for (int t = 0; t < 8; ++t) { float xv = z[t][v]; s += xv; sq += xv * xv; }
        s  += __shfl_xor(s, 1, 32);  s  += __shfl_xor(s, 2, 32);
        s  += __shfl_xor(s, 4, 32);  s  += __shfl_xor(s, 8, 32);
        sq += __shfl_xor(sq, 1, 32); sq += __shfl_xor(sq, 2, 32);
        sq += __shfl_xor(sq, 4, 32); sq += __shfl_xor(sq, 8, 32);
        mean[v] = s * (1.f / 128.f);
        rstd[v] = rsqrtf(sq * (1.f / 128.f) - mean[v] * mean[v] + EPS);
    }
    #pragma unroll
    for (int t = 0; t < 8; ++t) {
        const int n = t * 16 + m;
        const float w = prew[n], bb = preb[n];
        #pragma unroll
        for (int v = 0; v < 8; ++v) {
            float xv = (z[t][v] - mean[v]) * rstd[v] * w + bb;
            z[t][v] = xv;                         // keep residual in registers
            xt[v + g * 8][n] = (_Float16)xv;      // C/D layout -> row-major LDS
        }
    }
    lds_fence();
    // h = gelu(z @ W1 + b1), 256-wide
    #pragma unroll
    for (int t = 0; t < 16; ++t) {
        const int n = t * 16 + m;
        f8 acc = {};
        const _Float16* wr = W1T + (long)n * DIM;
        #pragma unroll
        for (int kc = 0; kc < 4; ++kc)
            acc = wmma16(load_a16(&xt[m][kc * 32], g), load_b16(wr + kc * 32 + g * 16), acc);
        const float bn = b1[n];
        #pragma unroll
        for (int v = 0; v < 8; ++v) {
            float xv = acc[v] + bn;
            xv = 0.5f * xv * (1.0f + erff(xv * 0.70710678118654752f));  // exact gelu
            ht[v + g * 8][n] = (_Float16)xv;
        }
    }
    lds_fence();
    // y = h @ W2 + b2 + z ; postnorm ; transposed store
    f8 y[8];
    #pragma unroll
    for (int t = 0; t < 8; ++t) {
        const int n = t * 16 + m;
        f8 acc = {};
        const _Float16* wr = W2T + (long)n * 256;
        #pragma unroll
        for (int kc = 0; kc < 8; ++kc)
            acc = wmma16(load_a16(&ht[m][kc * 32], g), load_b16(wr + kc * 32 + g * 16), acc);
        const float bn = b2[n];
        #pragma unroll
        for (int v = 0; v < 8; ++v) acc[v] += bn + z[t][v];
        y[t] = acc;
    }
    #pragma unroll
    for (int v = 0; v < 8; ++v) {
        float s = 0.f, sq = 0.f;
        #pragma unroll
        for (int t = 0; t < 8; ++t) { float xv = y[t][v]; s += xv; sq += xv * xv; }
        s  += __shfl_xor(s, 1, 32);  s  += __shfl_xor(s, 2, 32);
        s  += __shfl_xor(s, 4, 32);  s  += __shfl_xor(s, 8, 32);
        sq += __shfl_xor(sq, 1, 32); sq += __shfl_xor(sq, 2, 32);
        sq += __shfl_xor(sq, 4, 32); sq += __shfl_xor(sq, 8, 32);
        mean[v] = s * (1.f / 128.f);
        rstd[v] = rsqrtf(sq * (1.f / 128.f) - mean[v] * mean[v] + EPS);
    }
    #pragma unroll
    for (int t = 0; t < 8; ++t) {
        const int n = t * 16 + m;
        const float w = postw[n], bb = postb[n];
        #pragma unroll
        for (int v = 0; v < 8; ++v) {
            const int tok = tile * 16 + v + g * 8;
            if (tok < LQ)
                outp[((long)b * DIM + n) * LQ + tok] =
                    (y[t][v] - mean[v]) * rstd[v] * w + bb;
        }
    }
}

// ---------------- workspace layout (bytes, all 256-aligned) ----------------
constexpr size_t OFF_WQT = 0;
constexpr size_t OFF_WKT = OFF_WQT + (size_t)128 * 128 * 2;
constexpr size_t OFF_WVT = OFF_WKT + (size_t)128 * 128 * 2;
constexpr size_t OFF_WPT = OFF_WVT + (size_t)128 * 128 * 2;
constexpr size_t OFF_W1T = OFF_WPT + (size_t)128 * 128 * 2;
constexpr size_t OFF_W2T = OFF_W1T + (size_t)256 * 128 * 2;
constexpr size_t OFF_QH  = OFF_W2T + (size_t)128 * 256 * 2;
constexpr size_t OFF_KH  = OFF_QH + (size_t)2 * HEADS * LQP * DH * 2;
constexpr size_t OFF_VT  = OFF_KH + (size_t)2 * HEADS * LK * DH * 2;
constexpr size_t OFF_A   = OFF_VT + (size_t)2 * HEADS * LK * DH * 2;
constexpr size_t OFF_PO  = OFF_A  + (size_t)2 * LQP * DIM * 4;
constexpr size_t OFF_PM  = OFF_PO + (size_t)NSPLIT * 2 * HEADS * LQP * DH * 4;
constexpr size_t OFF_PL  = OFF_PM + (size_t)NSPLIT * 2 * HEADS * LQP * 4;

extern "C" void kernel_launch(void* const* d_in, const int* in_sizes, int n_in,
                              void* d_out, int out_size, void* d_ws, size_t ws_size,
                              hipStream_t stream) {
    (void)in_sizes; (void)n_in; (void)out_size; (void)ws_size;
    const float* q      = (const float*)d_in[0];
    const float* k      = (const float*)d_in[1];
    const float* v      = (const float*)d_in[2];
    const float* ln_q_w = (const float*)d_in[3];
    const float* ln_q_b = (const float*)d_in[4];
    const float* ln_k_w = (const float*)d_in[5];
    const float* ln_k_b = (const float*)d_in[6];
    const float* ln_v_w = (const float*)d_in[7];
    const float* ln_v_b = (const float*)d_in[8];
    const float* pre_w  = (const float*)d_in[9];
    const float* pre_b  = (const float*)d_in[10];
    const float* post_w = (const float*)d_in[11];
    const float* post_b = (const float*)d_in[12];
    const float* Wq = (const float*)d_in[13]; const float* bq = (const float*)d_in[14];
    const float* Wk = (const float*)d_in[15]; const float* bk = (const float*)d_in[16];
    const float* Wv = (const float*)d_in[17]; const float* bv = (const float*)d_in[18];
    const float* Wp = (const float*)d_in[19]; const float* bp = (const float*)d_in[20];
    const float* W1 = (const float*)d_in[21]; const float* b1 = (const float*)d_in[22];
    const float* W2 = (const float*)d_in[23]; const float* b2 = (const float*)d_in[24];
    float* out = (float*)d_out;

    char* ws = (char*)d_ws;
    _Float16* wqT = (_Float16*)(ws + OFF_WQT);
    _Float16* wkT = (_Float16*)(ws + OFF_WKT);
    _Float16* wvT = (_Float16*)(ws + OFF_WVT);
    _Float16* wpT = (_Float16*)(ws + OFF_WPT);
    _Float16* w1T = (_Float16*)(ws + OFF_W1T);
    _Float16* w2T = (_Float16*)(ws + OFF_W2T);
    _Float16* qhB = (_Float16*)(ws + OFF_QH);
    _Float16* khB = (_Float16*)(ws + OFF_KH);
    _Float16* vtB = (_Float16*)(ws + OFF_VT);
    float*    aB  = (float*)(ws + OFF_A);
    float*    poB = (float*)(ws + OFF_PO);
    float*    pmB = (float*)(ws + OFF_PM);
    float*    plB = (float*)(ws + OFF_PL);

    prep_weights<<<(128 * 128 + 255) / 256, 256, 0, stream>>>(Wq, wqT, 128, 128);
    prep_weights<<<(128 * 128 + 255) / 256, 256, 0, stream>>>(Wk, wkT, 128, 128);
    prep_weights<<<(128 * 128 + 255) / 256, 256, 0, stream>>>(Wv, wvT, 128, 128);
    prep_weights<<<(128 * 128 + 255) / 256, 256, 0, stream>>>(Wp, wpT, 128, 128);
    prep_weights<<<(128 * 256 + 255) / 256, 256, 0, stream>>>(W1, w1T, 128, 256);
    prep_weights<<<(256 * 128 + 255) / 256, 256, 0, stream>>>(W2, w2T, 256, 128);

    ln_proj<<<dim3(LQP / 16, 2), 32, 0, stream>>>(q, ln_q_w, ln_q_b, wqT, bq, qhB, LQ, LQP, 0);
    ln_proj<<<dim3(LK / 16, 2),  32, 0, stream>>>(k, ln_k_w, ln_k_b, wkT, bk, khB, LK, LK, 1);
    ln_proj<<<dim3(LK / 16, 2),  32, 0, stream>>>(v, ln_v_w, ln_v_b, wvT, bv, vtB, LK, LK, 2);

    attn<<<dim3(LQP / 16, HEADS, 2 * NSPLIT), 32, 0, stream>>>(qhB, khB, vtB, poB, pmB, plB);
    combine<<<(2 * LQP * DIM + 255) / 256, 256, 0, stream>>>(poB, pmB, plB, aB);

    ffn<<<dim3(LQP / 16, 2), 32, 0, stream>>>(aB, wpT, bp, w1T, b1, w2T, b2,
                                              pre_w, pre_b, post_w, post_b, out);
}
